// AttnDecoderRNN_38714835206244
// MI455X (gfx1250) — compile-verified, matrix operand order checked by reading, not run on hardware
//
#include <hip/hip_runtime.h>
#include <hip/hip_bf16.h>

#define Bdim 64
#define Sdim 512
#define Hdim 1024
#define Vdim 32000
#define LDA  (Hdim + 8)   // LDS row pitch in bf16 elems: 8-half pad -> bank spread for b128 reads

typedef __attribute__((ext_vector_type(16))) __bf16 v16bf;
typedef __attribute__((ext_vector_type(8)))  __bf16 v8bf;
typedef __attribute__((ext_vector_type(4)))  __bf16 v4bf;
typedef __attribute__((ext_vector_type(8)))  float  v8f;

__device__ __forceinline__ v16bf cat8(v8bf lo, v8bf hi) {
  return __builtin_shufflevector(lo, hi, 0,1,2,3,4,5,6,7,8,9,10,11,12,13,14,15);
}
__device__ __forceinline__ float sigmoidf(float x) { return 1.0f / (1.0f + __expf(-x)); }

// ---------------------------------------------------------------------------
// Prep: convert the encoder half of attn_w_W (columns H..2H) to bf16 [h][k].
__global__ void convert_wenc_kernel(const float* __restrict__ Waw, __bf16* __restrict__ Wenc) {
  int i = blockIdx.x * 256 + threadIdx.x;          // H*H = 1M
  int h = i >> 10, k = i & 1023;
  Wenc[i] = (__bf16)Waw[(size_t)h * (2 * Hdim) + Hdim + k];
}

// Embedding gather: embedded[b][h] = emb[ids[b]][h]
__global__ void embed_kernel(const int* __restrict__ ids, const float* __restrict__ emb,
                             float* __restrict__ embedded) {
  int i = blockIdx.x * 256 + threadIdx.x;          // B*H = 65536
  int b = i >> 10, h = i & 1023;
  embedded[i] = emb[(size_t)ids[b] * Hdim + h];
}

// hidproj_t[h][b] = attn_w_b[h] + hidden[b,:] . attn_w_W[h, 0:H]   (bias folded in)
__global__ void hidproj_kernel(const float* __restrict__ hidden, const float* __restrict__ Waw,
                               const float* __restrict__ ab, float* __restrict__ hidproj_t) {
  int b = blockIdx.x;                 // 64
  int h = threadIdx.x;                // 1024
  const float* x = &hidden[(size_t)b * Hdim];
  const float* w = &Waw[(size_t)h * (2 * Hdim)];
  float acc = ab[h];
  for (int k = 0; k < Hdim; ++k) acc += x[k] * w[k];
  hidproj_t[(size_t)h * Bdim + b] = acc;
}

// ---------------------------------------------------------------------------
// Big fused GEMM: scores[m] = sum_h v[h]*tanh(hidproj[b][h] + enc[m,:].Wenc[h,:])
// 512 WGs x 128 threads (4 waves). WG stages 64 enc rows in LDS as bf16 (read
// from HBM exactly once); each wave owns 16 M-rows, sweeps all N (h) tiles.
// Wenc (2MB bf16) is L2-resident on MI455X's 192MB L2. K loop is manually
// software-pipelined so fragment loads for chunk k+1 overlap the WMMA of
// chunk k (waitcnt becomes per-register, not a full drain).
__global__ void attn_scores_kernel(const float* __restrict__ enc,
                                   const __bf16* __restrict__ Wenc,
                                   const float* __restrict__ hidproj_t,
                                   const float* __restrict__ attn_v,
                                   float* __restrict__ scores) {
  extern __shared__ __bf16 smem[];   // 64 rows * LDA bf16 = 132096 B
  const int tid   = threadIdx.x;
  const int mbase = blockIdx.x * 64;

  // Cooperative f32->bf16 staging, vectorized (float4 in, b64 to LDS).
  for (int i = tid; i < (64 * Hdim) / 4; i += 128) {
    int r = i >> 8;
    int k = (i & 255) * 4;
    float4 f = *(const float4*)&enc[(size_t)(mbase + r) * Hdim + k];
    v4bf o; o[0] = (__bf16)f.x; o[1] = (__bf16)f.y; o[2] = (__bf16)f.z; o[3] = (__bf16)f.w;
    *(v4bf*)&smem[(size_t)r * LDA + k] = o;
  }
  __syncthreads();

  const int wave = tid >> 5;
  const int lane = tid & 31;
  const int ln   = lane & 15;        // A: row index ; B/D: column index
  const int half = lane >> 4;        // K sub-range select / D row bank
  const int m0   = wave * 16;        // this wave's row tile inside the WG block

  const __bf16* Abase = &smem[(size_t)(m0 + ln) * LDA];
  const int bbase = m0 + half * 8;   // b = global_m & 63 ; mbase % 64 == 0

  float acc[8] = {0.f, 0.f, 0.f, 0.f, 0.f, 0.f, 0.f, 0.f};

  for (int nt = 0; nt < Hdim / 16; ++nt) {
    const int h = nt * 16 + ln;
    const __bf16* Bbase = &Wenc[(size_t)h * Hdim];

    // Launder the A offset: without this, LLVM fully unrolls the K loop,
    // hoists all 1KB/lane of nt-invariant A fragments out of the nt loop and
    // spills them to scratch. Opaque offset -> A is re-read from LDS each
    // N-tile, which is the intended (cheap) data path.
    int kopq = 0;
    asm volatile("" : "+v"(kopq));
    const __bf16* Ab = Abase + kopq;

    v8f cacc = {};

    // Software-pipelined K loop: prefetch chunk k+1 while WMMA consumes k.
    int kk = half * 8;
    v8bf a0 = *(const v8bf*)(Ab + kk);
    v8bf a1 = *(const v8bf*)(Ab + kk + 16);
    v8bf b0 = *(const v8bf*)(Bbase + kk);
    v8bf b1 = *(const v8bf*)(Bbase + kk + 16);
#pragma unroll 2
    for (int kc = 32; kc < Hdim; kc += 32) {
      const int kn = kc + half * 8;
      v8bf na0 = *(const v8bf*)(Ab + kn);
      v8bf na1 = *(const v8bf*)(Ab + kn + 16);
      v8bf nb0 = *(const v8bf*)(Bbase + kn);
      v8bf nb1 = *(const v8bf*)(Bbase + kn + 16);
      cacc = __builtin_amdgcn_wmma_f32_16x16x32_bf16(false, cat8(a0, a1), false, cat8(b0, b1),
                                                     (short)0, cacc, false, false);
      a0 = na0; a1 = na1; b0 = nb0; b1 = nb1;
    }
    cacc = __builtin_amdgcn_wmma_f32_16x16x32_bf16(false, cat8(a0, a1), false, cat8(b0, b1),
                                                   (short)0, cacc, false, false);

    // Fused epilogue: e = tanh(gemm + hidproj); partial score += v[h]*e
    const float vv = attn_v[h];
    const float* hp = &hidproj_t[(size_t)h * Bdim + bbase];
#pragma unroll
    for (int r = 0; r < 8; ++r)
      acc[r] += vv * tanhf(cacc[r] + hp[r]);
  }

  // Reduce over the 16 column-lanes (h was striped over lanes within a half).
#pragma unroll
  for (int r = 0; r < 8; ++r) {
    float vsum = acc[r];
    vsum += __shfl_xor(vsum, 1, 32);
    vsum += __shfl_xor(vsum, 2, 32);
    vsum += __shfl_xor(vsum, 4, 32);
    vsum += __shfl_xor(vsum, 8, 32);
    acc[r] = vsum;
  }
  if (ln == 0) {
    const int mg = mbase + m0 + half * 8;
#pragma unroll
    for (int r = 0; r < 8; ++r) scores[mg + r] = acc[r];
  }
}

// ---------------------------------------------------------------------------
// Softmax over the S dimension per batch column. (attn_v_b is a per-column
// constant and cancels in softmax, so it is intentionally omitted.)
__global__ void softmax_kernel(const float* __restrict__ scores, float* __restrict__ attn_out) {
  __shared__ float red[256];
  const int b = blockIdx.x, t = threadIdx.x;
  float x0 = scores[t * Bdim + b];
  float x1 = scores[(t + 256) * Bdim + b];
  red[t] = fmaxf(x0, x1);
  __syncthreads();
  for (int s = 128; s > 0; s >>= 1) { if (t < s) red[t] = fmaxf(red[t], red[t + s]); __syncthreads(); }
  const float m = red[0];
  __syncthreads();
  const float e0 = __expf(x0 - m), e1 = __expf(x1 - m);
  red[t] = e0 + e1;
  __syncthreads();
  for (int s = 128; s > 0; s >>= 1) { if (t < s) red[t] += red[t + s]; __syncthreads(); }
  const float inv = 1.0f / red[0];
  attn_out[t * Bdim + b]        = e0 * inv;
  attn_out[(t + 256) * Bdim + b] = e1 * inv;
}

// contexts[b][h] = sum_s attn[s][b] * enc[s][b][h]   (second, final enc pass)
__global__ void context_kernel(const float* __restrict__ attn, const float* __restrict__ enc,
                               float* __restrict__ ctx) {
  const int b = blockIdx.x;
  const int h = blockIdx.y * 256 + threadIdx.x;
  float acc = 0.f;
  for (int s = 0; s < Sdim; ++s)
    acc += attn[s * Bdim + b] * enc[((size_t)s * Bdim + b) * Hdim + h];
  ctx[(size_t)b * Hdim + h] = acc;
}

// ---------------------------------------------------------------------------
// GRU single step, gates fused per (b,h). 0.6 GMAC; weights are L2-resident.
__global__ void gru_kernel(const float* __restrict__ embedded, const float* __restrict__ ctx,
                           const float* __restrict__ hidden,
                           const float* __restrict__ Wih, const float* __restrict__ Whh,
                           const float* __restrict__ bih, const float* __restrict__ bhh,
                           float* __restrict__ hnew_out, __bf16* __restrict__ hnew_bf) {
  const int b = blockIdx.x;
  const int h = blockIdx.y * 256 + threadIdx.x;
  const float* xe = &embedded[(size_t)b * Hdim];
  const float* xc = &ctx[(size_t)b * Hdim];
  const float* hp = &hidden[(size_t)b * Hdim];

  float gi_r = bih[h], gi_z = bih[Hdim + h], gi_n = bih[2 * Hdim + h];
  const float* wr = &Wih[(size_t)h * (2 * Hdim)];
  const float* wz = &Wih[(size_t)(Hdim + h) * (2 * Hdim)];
  const float* wn = &Wih[(size_t)(2 * Hdim + h) * (2 * Hdim)];
  for (int k = 0; k < Hdim; ++k) { float x = xe[k]; gi_r += x * wr[k]; gi_z += x * wz[k]; gi_n += x * wn[k]; }
  for (int k = 0; k < Hdim; ++k) { float x = xc[k]; gi_r += x * wr[Hdim + k]; gi_z += x * wz[Hdim + k]; gi_n += x * wn[Hdim + k]; }

  float gh_r = bhh[h], gh_z = bhh[Hdim + h], gh_n = bhh[2 * Hdim + h];
  const float* ur = &Whh[(size_t)h * Hdim];
  const float* uz = &Whh[(size_t)(Hdim + h) * Hdim];
  const float* un = &Whh[(size_t)(2 * Hdim + h) * Hdim];
  for (int k = 0; k < Hdim; ++k) { float x = hp[k]; gh_r += x * ur[k]; gh_z += x * uz[k]; gh_n += x * un[k]; }

  const float r = sigmoidf(gi_r + gh_r);
  const float z = sigmoidf(gi_z + gh_z);
  const float n = tanhf(gi_n + r * gh_n);
  const float hn = (1.0f - z) * n + z * hp[h];
  hnew_out[(size_t)b * Hdim + h] = hn;
  hnew_bf[(size_t)b * Hdim + h]  = (__bf16)hn;
}

// ---------------------------------------------------------------------------
// logits[b][v] = h_new . out_W[v,:] + out_b[v]. Memory bound on out_W (131MB):
// each WG covers ALL of M=64 for its 128 columns, so out_W streams exactly once.
// f32 rows converted to bf16 in VALU feeding the 16x16x32 WMMA pipe.
__global__ void logits_kernel(const __bf16* __restrict__ hnew_bf, const float* __restrict__ outW,
                              const float* __restrict__ outb, float* __restrict__ logits) {
  const int tid  = threadIdx.x;
  const int wave = tid >> 5, lane = tid & 31;
  const int ln   = lane & 15, half = lane >> 4;
  const int v    = blockIdx.x * 128 + wave * 16 + ln;
  const float* Wr = &outW[(size_t)v * Hdim];

  const v8f z = {};
  v8f cacc[4] = {z, z, z, z};

#pragma unroll 2
  for (int kc = 0; kc < Hdim; kc += 32) {
    const int kk = kc + half * 8;
    float4 w0 = *(const float4*)(Wr + kk);
    float4 w1 = *(const float4*)(Wr + kk + 4);
    float4 w2 = *(const float4*)(Wr + kk + 16);
    float4 w3 = *(const float4*)(Wr + kk + 20);
    v16bf Bm;
    Bm[0]  = (__bf16)w0.x; Bm[1]  = (__bf16)w0.y; Bm[2]  = (__bf16)w0.z; Bm[3]  = (__bf16)w0.w;
    Bm[4]  = (__bf16)w1.x; Bm[5]  = (__bf16)w1.y; Bm[6]  = (__bf16)w1.z; Bm[7]  = (__bf16)w1.w;
    Bm[8]  = (__bf16)w2.x; Bm[9]  = (__bf16)w2.y; Bm[10] = (__bf16)w2.z; Bm[11] = (__bf16)w2.w;
    Bm[12] = (__bf16)w3.x; Bm[13] = (__bf16)w3.y; Bm[14] = (__bf16)w3.z; Bm[15] = (__bf16)w3.w;
#pragma unroll
    for (int mt = 0; mt < 4; ++mt) {
      const __bf16* Ar = &hnew_bf[(size_t)(mt * 16 + ln) * Hdim + kk];
      v16bf Am = cat8(*(const v8bf*)Ar, *(const v8bf*)(Ar + 16));
      cacc[mt] = __builtin_amdgcn_wmma_f32_16x16x32_bf16(false, Am, false, Bm,
                                                         (short)0, cacc[mt], false, false);
    }
  }

  const float ob = outb[v];
#pragma unroll
  for (int mt = 0; mt < 4; ++mt)
#pragma unroll
    for (int r = 0; r < 8; ++r) {
      const int brow = mt * 16 + half * 8 + r;
      logits[(size_t)brow * Vdim + v] = cacc[mt][r] + ob;
    }
}

// ---------------------------------------------------------------------------
extern "C" void kernel_launch(void* const* d_in, const int* in_sizes, int n_in,
                              void* d_out, int out_size, void* d_ws, size_t ws_size,
                              hipStream_t stream) {
  const int*   ids    = (const int*)  d_in[0];
  const float* hidden = (const float*)d_in[1];
  const float* enc    = (const float*)d_in[2];
  const float* emb    = (const float*)d_in[3];
  const float* attnW  = (const float*)d_in[4];
  const float* attnb  = (const float*)d_in[5];
  const float* attnv  = (const float*)d_in[6];
  /* d_in[7] attn_v_b: constant per softmax column -> cancels, unused */
  const float* Wih    = (const float*)d_in[8];
  const float* Whh    = (const float*)d_in[9];
  const float* bih    = (const float*)d_in[10];
  const float* bhh    = (const float*)d_in[11];
  const float* outW   = (const float*)d_in[12];
  const float* outb   = (const float*)d_in[13];

  float* out_logits = (float*)d_out;                              // [B,V]
  float* out_hidden = out_logits + (size_t)Bdim * Vdim;           // [1,B,H]
  float* out_attn   = out_hidden + (size_t)Bdim * Hdim;           // [S,B,1]

  // Workspace carve-up (≈3 MB, all 16B-aligned offsets)
  char* wsb = (char*)d_ws;
  __bf16* Wenc      = (__bf16*)(wsb);                             // H*H bf16   = 2 MB
  float*  hidproj_t = (float*) (wsb + 2097152);                   // H*B f32    = 256 KB
  float*  embedded  = (float*) (wsb + 2097152 + 262144);          // B*H f32    = 256 KB
  float*  scores    = (float*) (wsb + 2097152 + 524288);          // S*B f32    = 128 KB
  float*  ctx       = (float*) (wsb + 2097152 + 655360);          // B*H f32    = 256 KB
  __bf16* hnewbf    = (__bf16*)(wsb + 2097152 + 917504);          // B*H bf16   = 128 KB

  const int attn_lds = 64 * LDA * sizeof(__bf16);                 // 132096 B dynamic LDS
  (void)hipFuncSetAttribute((const void*)attn_scores_kernel,
                            hipFuncAttributeMaxDynamicSharedMemorySize, attn_lds);

  convert_wenc_kernel<<<(Hdim * Hdim) / 256, 256, 0, stream>>>(attnW, Wenc);
  embed_kernel<<<(Bdim * Hdim) / 256, 256, 0, stream>>>(ids, emb, embedded);
  hidproj_kernel<<<Bdim, Hdim, 0, stream>>>(hidden, attnW, attnb, hidproj_t);
  attn_scores_kernel<<<(Sdim * Bdim) / 64, 128, attn_lds, stream>>>(enc, Wenc, hidproj_t,
                                                                    attnv, scores);
  softmax_kernel<<<Bdim, 256, 0, stream>>>(scores, out_attn);
  context_kernel<<<dim3(Bdim, Hdim / 256), 256, 0, stream>>>(out_attn, enc, ctx);
  gru_kernel<<<dim3(Bdim, Hdim / 256), 256, 0, stream>>>(embedded, ctx, hidden, Wih, Whh,
                                                         bih, bhh, out_hidden, hnewbf);
  logits_kernel<<<Vdim / 128, 256, 0, stream>>>(hnewbf, outW, outb, out_logits);
}
